// MDTA_55482387530511
// MI455X (gfx1250) — matrix operand order repaired and checked by reading, not compiled, not run on hardware
//
#include <hip/hip_runtime.h>
#include <math.h>

// ---------------------------------------------------------------------------
// MDTA (Restormer channel attention) for MI455X / gfx1250, wave32 + WMMA.
// B=4, C=64, H=W=256, NH=4, d=16, N=HW=65536.
//
// Structure (memory-bound problem, ~10.5 GFLOP vs 23.3 TB/s HBM):
//   k_pointwise : 192x64x(B*HW) GEMM -> fp16 t (96MB, L2-resident)
//   k_dw_v      : depthwise 3x3 on v channels -> fp16 v (32MB)
//   k_gram      : depthwise on q,k + WMMA Gram/norm global reductions
//   k_attn      : softmax + fold proj_w into one 64x64 matrix M[b]
//   k_out       : 64x64x(HW) GEMM  out = M[b] @ v
// ---------------------------------------------------------------------------

typedef __attribute__((ext_vector_type(16))) _Float16 v16h;
typedef __attribute__((ext_vector_type(8)))  _Float16 v8h;
typedef __attribute__((ext_vector_type(8)))  float    v8f;
typedef __attribute__((ext_vector_type(4)))  int      v4i;

#define HWN   65536   // H*W
#define IMGW  256
#define CCH   64
#define NHEAD 4

#ifndef __has_builtin
#define __has_builtin(x) 0
#endif
#if __has_builtin(__builtin_amdgcn_global_load_async_to_lds_b128)
#define HAVE_ASYNC_COPY 1
#else
#define HAVE_ASYNC_COPY 0
#endif

#if HAVE_ASYNC_COPY
// global (AS1) -> LDS (AS3), 16 bytes per lane, tracked by ASYNCcnt
__device__ __forceinline__ void async_copy_b128(const void* gsrc, void* lds) {
  __builtin_amdgcn_global_load_async_to_lds_b128(
      (__attribute__((address_space(1))) v4i*)gsrc,
      (__attribute__((address_space(3))) v4i*)lds, 0, 0);
}
#endif

__device__ __forceinline__ v8f wmma_f16(v16h a, v16h b, v8f c) {
  // D = A(16x32 f16) * B(32x16 f16) + C(16x16 f32)
  return __builtin_amdgcn_wmma_f32_16x16x32_f16(
      false, a, false, b, (short)0, c, false, false);
}

// A-matrix 16x32 f16 fragment from a row-major [16][ld] f16 tile.
// Contiguous k order: lanes 0-15 row M=lane hold K={0..15}; lanes 16-31 K={16..31}.
// Halves are contiguous in runs of 16 -> compiler emits ds_load_b128 pairs.
__device__ __forceinline__ v16h load_a_frag(const _Float16* tile, int ld, int lane) {
  v16h a;
  int m = lane & 15;
  int koff = (lane >> 4) << 4;
#pragma unroll
  for (int i = 0; i < 16; ++i) {
    int k = i + koff;
    a[i] = tile[m * ld + k];
  }
  return a;
}

// B-matrix 32x16 f16 fragment; source tile stored [N=16 rows][K=32 cols].
__device__ __forceinline__ v16h load_b_frag(const _Float16* tile, int ld, int lane) {
  v16h b;
  int n = lane & 15;
  int koff = (lane >> 4) << 4;
#pragma unroll
  for (int i = 0; i < 16; ++i) b[i] = tile[n * ld + koff + i];
  return b;
}

// ---------------------------------------------------------------------------
// Kernel 1: pointwise qkv conv  t[b][o][p] = sum_c qkv_w[o][c] * x[b][c][p]
// Block (8 waves) covers 128 contiguous pixels; x tile staged in LDS with
// coalesced float4 loads; 16x16 D tiles repacked through LDS so the fp16
// stores go out as full b128 / 256B-per-row transactions.
// ---------------------------------------------------------------------------
__global__ __launch_bounds__(256) void k_pointwise(
    const float* __restrict__ x, const float* __restrict__ qkv_w,
    _Float16* __restrict__ t) {
  __shared__ _Float16 sW[192 * 64];    // 24 KB  weights (f16)
  __shared__ _Float16 sX[64 * 128];    // 16 KB  x tile (f16)
  __shared__ _Float16 sO[16 * 128];    //  4 KB  output repack tile
  int tid = threadIdx.x;
  for (int i = tid; i < 192 * 64; i += 256) sW[i] = (_Float16)qkv_w[i];

  long n0 = (long)blockIdx.x * 128;           // 2048 blocks x 128 px
  int b  = (int)(n0 >> 16);
  int p0 = (int)(n0 & (HWN - 1));
  const float* xb = x + (long)b * CCH * HWN;

  // stage x tile [64 ch][128 px] as f16, coalesced float4 reads
#pragma unroll
  for (int j = 0; j < 8; ++j) {
    int idx = tid + j * 256;          // 2048 float4 elements
    int c   = idx >> 5;               // 32 float4 per row
    int c4  = (idx & 31) << 2;
    float4 f = *(const float4*)(xb + ((long)c << 16) + p0 + c4);
    sX[c * 128 + c4 + 0] = (_Float16)f.x;
    sX[c * 128 + c4 + 1] = (_Float16)f.y;
    sX[c * 128 + c4 + 2] = (_Float16)f.z;
    sX[c * 128 + c4 + 3] = (_Float16)f.w;
  }
  __syncthreads();

  int wave = tid >> 5, lane = tid & 31;
  int n = lane & 15;
  int colw = wave * 16 + n;

  // B fragments (K = input channel, N = pixel) from LDS
  int krow = (lane >> 4) << 4;
  v16h b0, b1;
#pragma unroll
  for (int i = 0; i < 16; ++i) {
    b0[i] = sX[(krow + i) * 128 + colw];
    b1[i] = sX[(krow + i + 32) * 128 + colw];
  }

  int m = lane & 15;
  int koffA = (lane >> 4) << 4;
  int rbase = (lane >> 4) << 3;
  int srow = tid >> 4;                 // repack-store row (0..15)
  int sc0  = (tid & 15) << 3;          // repack-store col (8 halves)

  for (int mt = 0; mt < 12; ++mt) {    // 12 output-channel tiles of 16
    const _Float16* wrow = &sW[(mt * 16 + m) * 64];
    v16h a0, a1;
#pragma unroll
    for (int i = 0; i < 16; ++i) {
      int k = i + koffA;
      a0[i] = wrow[k];
      a1[i] = wrow[k + 32];
    }
    v8f d = {};
    d = wmma_f16(a0, b0, d);
    d = wmma_f16(a1, b1, d);
#pragma unroll
    for (int r = 0; r < 8; ++r)        // D: m = r + rbase, n = lane&15
      sO[(r + rbase) * 128 + colw] = (_Float16)d[r];
    __syncthreads();
    // coalesced store: 16 rows x 256B, one b128 per thread
    int o = mt * 16 + srow;
    *(v8h*)(t + (((long)b * 192 + o) << 16) + p0 + sc0) =
        *(const v8h*)&sO[srow * 128 + sc0];
    __syncthreads();
  }
}

// ---------------------------------------------------------------------------
// Kernel 2a: depthwise 3x3 for the v channels (128..191) -> fp16 v[b][c][p]
// ---------------------------------------------------------------------------
__global__ __launch_bounds__(256) void k_dw_v(
    const _Float16* __restrict__ t, const float* __restrict__ dw,
    _Float16* __restrict__ v) {
  int b  = blockIdx.x >> 6;
  int cl = blockIdx.x & 63;
  int c  = 128 + cl;
  const _Float16* src = t + ((long)b * 192 + c) * HWN;
  _Float16* dst = v + ((long)b * CCH + cl) * HWN;
  float w[9];
#pragma unroll
  for (int i = 0; i < 9; ++i) w[i] = dw[c * 9 + i];
  for (int p = threadIdx.x; p < HWN; p += 256) {
    if (p + 512 < HWN) __builtin_prefetch(src + p + 512, 0, 0);  // global_prefetch_b8
    int y = p >> 8, xx = p & 255;
    float acc = 0.f;
#pragma unroll
    for (int ky = 0; ky < 3; ++ky) {
      int yy = y + ky - 1;
      if (yy < 0 || yy > 255) continue;
#pragma unroll
      for (int kx = 0; kx < 3; ++kx) {
        int xc = xx + kx - 1;
        if (xc < 0 || xc > 255) continue;
        acc += w[ky * 3 + kx] * (float)src[yy * IMGW + xc];
      }
    }
    dst[p] = (_Float16)acc;
  }
}

// ---------------------------------------------------------------------------
// Kernel 2b: depthwise for q,k channels + Gram/norm reductions via WMMA.
// Block = 8 waves handles (b, head, 8 image rows). 10 halo rows x 32
// channels staged in dynamic LDS via async global->LDS copy (one b128
// wave-op per 512B image row), tracked by ASYNCcnt. Per 32-pixel chunk:
// 3 WMMAs (q*k^T, q*q^T, k*k^T); diagonals give the L2 norms for free.
// ---------------------------------------------------------------------------
__global__ __launch_bounds__(256) void k_gram(
    const _Float16* __restrict__ t, const float* __restrict__ dw,
    float* __restrict__ G, float* __restrict__ qq, float* __restrict__ kk) {
  extern __shared__ _Float16 smem[];
  _Float16* s_t     = smem;                       // [2][16][10][256]
  _Float16* s_stage = smem + 2 * 16 * 10 * 256;   // [8 waves][2][16][32]

  int blk = blockIdx.x;                 // 512 = b(4) * h(4) * ygroup(32)
  int yg = blk & 31;
  int h  = (blk >> 5) & 3;
  int b  = blk >> 7;
  int y0 = yg * 8;
  int tid = threadIdx.x;
  int wave = tid >> 5, lane = tid & 31;

#if HAVE_ASYNC_COPY
  // rows y0-1 .. y0+8 of 32 channels: one async b128 per lane per row
  for (int r = wave; r < 320; r += 8) {
    int row = r % 10;
    int dq  = r / 10;                   // qk*16 + d
    int d   = dq & 15;
    int qk  = dq >> 4;
    int gy  = y0 - 1 + row;
    _Float16* ldsrow = s_t + r * 256 + lane * 8;
    if (gy >= 0 && gy < 256) {
      const _Float16* gsrc =
          t + (((long)b * 192 + qk * 64 + h * 16 + d) << 16) + (gy << 8) + lane * 8;
      async_copy_b128(gsrc, ldsrow);
    } else {
      v8h z = {};
      *(v8h*)ldsrow = z;                // zero-pad halo outside the image
    }
  }
#if __has_builtin(__builtin_amdgcn_s_wait_asynccnt)
  __builtin_amdgcn_s_wait_asynccnt(0);
#else
  asm volatile("s_wait_asynccnt 0" ::: "memory");
#endif
#else
  for (int e = tid; e < 2 * 16 * 10 * 256; e += 256) {
    int xx  = e & 255;
    int r2  = e >> 8;
    int row = r2 % 10;
    int dq  = r2 / 10;
    int d   = dq & 15;
    int qk  = dq >> 4;
    int gy  = y0 - 1 + row;
    _Float16 val = (_Float16)0.f;
    if (gy >= 0 && gy < 256)
      val = t[(((long)b * 192 + qk * 64 + h * 16 + d) << 16) + (gy << 8) + xx];
    s_t[e] = val;
  }
#endif
  __syncthreads();

  _Float16* stg = s_stage + wave * 2 * 16 * 32;
  v8f g = {}, dqq = {}, dkk = {};

  for (int cx = 0; cx < 8; ++cx) {      // 8 chunks of 32 pixels in this row
    int xx = cx * 32 + lane;
    for (int qk = 0; qk < 2; ++qk) {
#pragma unroll
      for (int d = 0; d < 16; ++d) {
        const float* wp = dw + (qk * 64 + h * 16 + d) * 9;   // scalar loads
        float acc = 0.f;
#pragma unroll
        for (int ky = 0; ky < 3; ++ky) {
          const _Float16* rowp = s_t + (((qk * 16 + d) * 10) + wave + ky) * 256;
#pragma unroll
          for (int kx = 0; kx < 3; ++kx) {
            int xc = xx + kx - 1;
            float val = (xc >= 0 && xc < 256) ? (float)rowp[xc] : 0.f;
            acc = fmaf(wp[ky * 3 + kx], val, acc);
          }
        }
        stg[(qk * 16 + d) * 32 + lane] = (_Float16)acc;      // [ch][px]
      }
    }
    asm volatile("s_wait_dscnt 0" ::: "memory");  // wave-local LDS RAW fence

    const _Float16* tq = stg;
    const _Float16* tk = stg + 16 * 32;
    v16h aq = load_a_frag(tq, 32, lane);
    v16h ak = load_a_frag(tk, 32, lane);
    v16h bq = load_b_frag(tq, 32, lane);
    v16h bk = load_b_frag(tk, 32, lane);
    g   = wmma_f16(aq, bk, g);     // Gram  q . k^T
    dqq = wmma_f16(aq, bq, dqq);   // diag -> sum q^2
    dkk = wmma_f16(ak, bk, dkk);   // diag -> sum k^2
  }

  // accumulate global reductions
  float* Gp = G + ((b * NHEAD + h) << 8);
  int rbase = (lane >> 4) << 3;
  int n = lane & 15;
#pragma unroll
  for (int r = 0; r < 8; ++r) {
    int m = r + rbase;
    atomicAdd(&Gp[m * 16 + n], g[r]);
    if (m == n) {
      atomicAdd(&qq[(b * NHEAD + h) * 16 + m], dqq[r]);
      atomicAdd(&kk[(b * NHEAD + h) * 16 + m], dkk[r]);
    }
  }
}

// ---------------------------------------------------------------------------
// Kernel 3: normalize + softmax + fold proj:  M[b] = proj_w @ blockdiag(attn)
// ---------------------------------------------------------------------------
__global__ __launch_bounds__(64) void k_attn(
    const float* __restrict__ G, const float* __restrict__ qq,
    const float* __restrict__ kk, const float* __restrict__ temp,
    const float* __restrict__ proj_w, _Float16* __restrict__ M) {
  int b = blockIdx.x;
  __shared__ float s_attn[NHEAD][16][16];
  __shared__ float s_nq[64], s_nk[64];
  int tid = threadIdx.x;                 // 64 threads = (h, d)

  s_nq[tid] = fmaxf(sqrtf(qq[b * 64 + tid]), 1e-12f);
  s_nk[tid] = fmaxf(sqrtf(kk[b * 64 + tid]), 1e-12f);
  __syncthreads();

  {
    int h = tid >> 4, d = tid & 15;
    float tmp = temp[h];
    float row[16], mx = -INFINITY;
#pragma unroll
    for (int e = 0; e < 16; ++e) {
      float val = G[(((b * NHEAD + h) * 16) + d) * 16 + e] /
                  (s_nq[h * 16 + d] * s_nk[h * 16 + e]) * tmp;
      row[e] = val;
      mx = fmaxf(mx, val);
    }
    float s = 0.f;
#pragma unroll
    for (int e = 0; e < 16; ++e) { row[e] = __expf(row[e] - mx); s += row[e]; }
    float inv = 1.f / s;
#pragma unroll
    for (int e = 0; e < 16; ++e) s_attn[h][d][e] = row[e] * inv;
  }
  __syncthreads();

  // M[b][o][c] = sum_d proj_w[o][h*16+d] * attn[h][d][e],  c = h*16+e
  for (int j = tid; j < 64 * 64; j += 64) {
    int o = j >> 6, c = j & 63, h2 = c >> 4, e = c & 15;
    float acc = 0.f;
#pragma unroll
    for (int d = 0; d < 16; ++d)
      acc += proj_w[o * 64 + h2 * 16 + d] * s_attn[h2][d][e];
    M[(b * 64 + o) * 64 + c] = (_Float16)acc;
  }
}

// ---------------------------------------------------------------------------
// Kernel 4: out[b][o][p] = sum_c M[b][o][c] * v[b][c][p]   (64x64xHW GEMM)
// v tile staged in LDS (coalesced b128), f32 output repacked through LDS
// so stores go out as 512B-per-row transactions.
// ---------------------------------------------------------------------------
__global__ __launch_bounds__(256) void k_out(
    const _Float16* __restrict__ v, const _Float16* __restrict__ M,
    float* __restrict__ out) {
  __shared__ _Float16 sM[64 * 64];     //  8 KB
  __shared__ _Float16 sV[64 * 128];    // 16 KB
  __shared__ float    sO[16 * 128];    //  8 KB
  int blk = blockIdx.x;                 // 2048 = b(4) * 512 groups of 128 px
  int b  = blk >> 9;
  int pg = blk & 511;
  int p0 = pg * 128;
  int tid = threadIdx.x;

  for (int i = tid; i < 64 * 64; i += 256) sM[i] = M[b * 4096 + i];
  const _Float16* vb = v + (long)b * CCH * HWN;
#pragma unroll
  for (int j = 0; j < 4; ++j) {         // 64x128 f16 tile, b128 coalesced
    int e  = tid + j * 256;             // 1024 v8h elements
    int c  = e >> 4;                    // 16 v8h per row
    int c0 = (e & 15) << 3;
    *(v8h*)&sV[c * 128 + c0] = *(const v8h*)(vb + ((long)c << 16) + p0 + c0);
  }
  __syncthreads();

  int wave = tid >> 5, lane = tid & 31;
  int n = lane & 15;
  int colw = wave * 16 + n;
  int koff = (lane >> 4) << 4;
  v16h b0, b1;
#pragma unroll
  for (int i = 0; i < 16; ++i) {
    b0[i] = sV[(koff + i) * 128 + colw];
    b1[i] = sV[(koff + i + 32) * 128 + colw];
  }

  int m = lane & 15;
  int koffA = (lane >> 4) << 4;
  int rbase = (lane >> 4) << 3;
  int srow = tid >> 4;
  int sc0  = (tid & 15) << 3;
  float* ob = out + (long)b * CCH * HWN;

  for (int ot = 0; ot < 4; ++ot) {
    const _Float16* mrow = &sM[(ot * 16 + m) * 64];
    v16h a0, a1;
#pragma unroll
    for (int i = 0; i < 16; ++i) {
      int k = i + koffA;
      a0[i] = mrow[k];
      a1[i] = mrow[k + 32];
    }
    v8f d = {};
    d = wmma_f16(a0, b0, d);
    d = wmma_f16(a1, b1, d);
#pragma unroll
    for (int r = 0; r < 8; ++r)
      sO[(r + rbase) * 128 + colw] = d[r];
    __syncthreads();
    int o = ot * 16 + srow;
    float* dst = ob + ((long)o << 16) + p0 + sc0;
    *(float4*)(dst + 0) = *(const float4*)&sO[srow * 128 + sc0];
    *(float4*)(dst + 4) = *(const float4*)&sO[srow * 128 + sc0 + 4];
    __syncthreads();
  }
}

// ---------------------------------------------------------------------------
extern "C" void kernel_launch(void* const* d_in, const int* in_sizes, int n_in,
                              void* d_out, int out_size, void* d_ws, size_t ws_size,
                              hipStream_t stream) {
  const float* x           = (const float*)d_in[0];
  const float* qkv_w       = (const float*)d_in[1];
  const float* qkv_dw_w    = (const float*)d_in[2];
  const float* proj_w      = (const float*)d_in[3];
  const float* temperature = (const float*)d_in[4];
  float* out = (float*)d_out;

  char* ws = (char*)d_ws;
  _Float16* t  = (_Float16*)ws;                          // 4*192*65536*2 = 96 MB
  _Float16* vv = (_Float16*)(ws + 100663296);            // 4*64*65536*2  = 32 MB
  float* G  = (float*)(ws + 100663296 + 33554432);       // 4*4*256 f32
  float* qq = G + 4096;                                  // 4*4*16
  float* kk = qq + 256;
  _Float16* M = (_Float16*)(kk + 256);                   // 4*64*64 f16

  (void)hipMemsetAsync(G, 0, (4096 + 256 + 256) * sizeof(float), stream);

  k_pointwise<<<2048, 256, 0, stream>>>(x, qkv_w, t);
  k_dw_v<<<256, 256, 0, stream>>>(t, qkv_dw_w, vv);

  size_t smem_gram = (size_t)(2 * 16 * 10 * 256 + 8 * 2 * 16 * 32) * sizeof(_Float16);
  (void)hipFuncSetAttribute((const void*)k_gram,
                            hipFuncAttributeMaxDynamicSharedMemorySize,
                            (int)smem_gram);
  k_gram<<<512, 256, smem_gram, stream>>>(t, qkv_dw_w, G, qq, kk);

  k_attn<<<4, 64, 0, stream>>>(G, qq, kk, temperature, proj_w, M);
  k_out<<<2048, 256, 0, stream>>>(vv, M, out);
}
